// Tacotron2Decoder_23167053595282
// MI455X (gfx1250) — matrix operand order repaired and checked by reading, compiled, not run on hardware
//
#include <hip/hip_runtime.h>
#include <hip/hip_bf16.h>
#include <stdint.h>

// ---------------------------------------------------------------------------
// Tacotron2 decoder for MI455X (gfx1250).  bf16 WMMA GEMMs everywhere a
// matrix multiply appears; f32 VALU for small pointwise/softmax work.
// Register-blocked WMMA (NB N-tiles per wave, 2 M-tiles in the conv GEMM),
// load-phase/compute-phase split so the scheduler can issue partial
// s_wait_loadcnt instead of full drains, global_prefetch on weight tiles.
// ---------------------------------------------------------------------------

typedef __attribute__((ext_vector_type(16))) __bf16 v16bf;
typedef __attribute__((ext_vector_type(8)))  __bf16 v8bf;
typedef __attribute__((ext_vector_type(8)))  float  v8f;

#define B_   64
#define TE_  512
#define TD_  500
#define MEL_ 80
#define PRE_ 256
#define ED_  512
#define H_   1024
#define ATT_ 128

static __device__ __forceinline__ __bf16 f2bf(float f) {
  union { float f; unsigned u; } a; a.f = f;
  unsigned r = a.u + 0x7FFFu + ((a.u >> 16) & 1u);
  union { unsigned short u; __bf16 h; } b; b.u = (unsigned short)(r >> 16);
  return b.h;
}
static __device__ __forceinline__ float sigm(float x) { return 1.f / (1.f + __expf(-x)); }

// ---- WMMA fragment loaders -------------------------------------------------
// A (16x32 bf16, row-major source): lane m=l&15, kh=l>>4.
//   elems 0..15 -> K = kbase + 16*kh + e          (contiguous 32B)
static __device__ __forceinline__ v16bf load_a_frag(const __bf16* A, int lda,
                                                    int row0, int kbase) {
  int l = threadIdx.x & 31;
  int m = l & 15, kh = l >> 4;
  const __bf16* p = A + (size_t)(row0 + m) * lda + kbase + 16 * kh;
  v8bf lo = *reinterpret_cast<const v8bf*>(p);
  v8bf hi = *reinterpret_cast<const v8bf*>(p + 8);
  v16bf r;
#pragma unroll
  for (int i = 0; i < 8; ++i) { r[i] = lo[i]; r[i + 8] = hi[i]; }
  return r;
}
// B (32x16 bf16) from swizzled weight store: per 16(n)x32(k) tile,
// 512 bf16 contiguous; lane l reads 16 contiguous values.
// swizzle wrote: [tile*512 + l*16 + e] = W^T[k = kt*32 + e + 16*(l>>4)][n = nt*16 + (l&15)]
static __device__ __forceinline__ v16bf load_b_frag(const __bf16* Bsw, size_t tile) {
  int l = threadIdx.x & 31;
  const __bf16* p = Bsw + tile * 512 + (size_t)l * 16;
  v8bf lo = *reinterpret_cast<const v8bf*>(p);
  v8bf hi = *reinterpret_cast<const v8bf*>(p + 8);
  v16bf r;
#pragma unroll
  for (int i = 0; i < 8; ++i) { r[i] = lo[i]; r[i + 8] = hi[i]; }
  return r;
}

// ---- generic bf16 WMMA GEMM: C[M,N] = A[M,K] @ Bsw + bias, act ------------
// grid = (Ntiles/NB, Mtiles), block = 32 (one wave, NB 16x16 tiles per wave)
template <int NB>
__global__ void __launch_bounds__(32)
gemm_ws(const __bf16* __restrict__ A, int lda, const __bf16* __restrict__ Bsw,
        const float* __restrict__ bias, float* __restrict__ Cf, int ldc,
        __bf16* __restrict__ Cb, int ldcb, int Ktiles, int Ntiles, int act) {
  int nt0 = blockIdx.x * NB, mt = blockIdx.y;
  int row0 = mt * 16;
  v8f acc[NB];
#pragma unroll
  for (int j = 0; j < NB; ++j) acc[j] = (v8f){};
  for (int kt = 0; kt < Ktiles; ++kt) {
    if (kt + 1 < Ktiles)
      __builtin_prefetch(Bsw + ((size_t)(kt + 1) * Ntiles + nt0) * 512, 0, 3);
    // load phase: A fragment + all NB B fragments in flight
    v16bf a = load_a_frag(A, lda, row0, kt * 32);
    v16bf bf[NB];
#pragma unroll
    for (int j = 0; j < NB; ++j)
      bf[j] = load_b_frag(Bsw, (size_t)kt * Ntiles + nt0 + j);
    // compute phase
#pragma unroll
    for (int j = 0; j < NB; ++j)
      acc[j] = __builtin_amdgcn_wmma_f32_16x16x32_bf16(false, a, false, bf[j],
                                                       (short)0, acc[j], false, false);
  }
  int l = threadIdx.x & 31;
  int nn = l & 15, kh = l >> 4;
#pragma unroll
  for (int j = 0; j < NB; ++j) {
    int col0 = (nt0 + j) * 16;
    float bv = bias ? bias[col0 + nn] : 0.f;
#pragma unroll
    for (int r = 0; r < 8; ++r) {
      int m = row0 + r + 8 * kh;
      float v = acc[j][r] + bv;
      if (act == 1) v = v > 0.f ? v : 0.f;
      if (Cf) Cf[(size_t)m * ldc + col0 + nn] = v;
      if (Cb) Cb[(size_t)m * ldcb + col0 + nn] = f2bf(v);
    }
  }
}

// ---- postnet conv as 5 fused shifted GEMMs --------------------------------
// X: bf16 [B, 520, lda] zero-padded in t (row = t+2).  Y: f32 [B, 512, ldy].
// grid = (Ntiles/NB, B*16); each wave owns 2 M-tiles x NB N-tiles.
template <int NB>
__global__ void __launch_bounds__(32)
gemm_conv(const __bf16* __restrict__ X, int lda, const __bf16* __restrict__ Wsw,
          const float* __restrict__ bias, float* __restrict__ Y, int ldy,
          int Ktiles, int Ntiles) {
  int nt0 = blockIdx.x * NB;
  int b = blockIdx.y >> 4, tp = blockIdx.y & 15;
  int arow0 = b * 520 + tp * 32;
  int crow0 = b * 512 + tp * 32;
  v8f acc[2][NB];
#pragma unroll
  for (int mi = 0; mi < 2; ++mi)
#pragma unroll
    for (int j = 0; j < NB; ++j) acc[mi][j] = (v8f){};
  for (int k = 0; k < 5; ++k)
    for (int kt = 0; kt < Ktiles; ++kt) {
      int nx = k * Ktiles + kt + 1;
      if (nx < 5 * Ktiles)
        __builtin_prefetch(Wsw + ((size_t)nx * Ntiles + nt0) * 512, 0, 3);
      // load phase
      v16bf a0 = load_a_frag(X, lda, arow0 + k, kt * 32);
      v16bf a1 = load_a_frag(X, lda, arow0 + 16 + k, kt * 32);
      v16bf wf[NB];
#pragma unroll
      for (int j = 0; j < NB; ++j)
        wf[j] = load_b_frag(Wsw, (size_t)(k * Ktiles + kt) * Ntiles + nt0 + j);
      // compute phase
#pragma unroll
      for (int j = 0; j < NB; ++j) {
        acc[0][j] = __builtin_amdgcn_wmma_f32_16x16x32_bf16(false, a0, false, wf[j],
                                                            (short)0, acc[0][j], false, false);
        acc[1][j] = __builtin_amdgcn_wmma_f32_16x16x32_bf16(false, a1, false, wf[j],
                                                            (short)0, acc[1][j], false, false);
      }
    }
  int l = threadIdx.x & 31;
  int nn = l & 15, kh = l >> 4;
#pragma unroll
  for (int mi = 0; mi < 2; ++mi)
#pragma unroll
    for (int j = 0; j < NB; ++j) {
      int col0 = (nt0 + j) * 16;
      float bv = bias[col0 + nn];
#pragma unroll
      for (int r = 0; r < 8; ++r)
        Y[(size_t)(crow0 + mi * 16 + r + 8 * kh) * ldy + col0 + nn] = acc[mi][j][r] + bv;
    }
}

// ---- weight converters (f32 -> fragment-swizzled bf16) --------------------
__global__ void wswz(const float* __restrict__ W, int N, int K, int sN, int sK,
                     __bf16* __restrict__ out, int Ktiles, int Ntiles) {
  size_t idx = (size_t)blockIdx.x * blockDim.x + threadIdx.x;
  size_t tot = (size_t)Ktiles * Ntiles * 512;
  if (idx >= tot) return;
  int e = idx & 15, lane = (idx >> 4) & 31;
  size_t tile = idx >> 9;
  int nt = (int)(tile % Ntiles), kt = (int)(tile / Ntiles);
  int n = nt * 16 + (lane & 15);
  int k = kt * 32 + e + 16 * (lane >> 4);
  float v = (n < N && k < K) ? W[(size_t)n * sN + (size_t)k * sK] : 0.f;
  out[idx] = f2bf(v);
}
__global__ void wswz_cat(const float* __restrict__ wih, const float* __restrict__ whh,
                         int Kih, int Khh, __bf16* __restrict__ out,
                         int Ktiles, int Ntiles) {
  size_t idx = (size_t)blockIdx.x * blockDim.x + threadIdx.x;
  size_t tot = (size_t)Ktiles * Ntiles * 512;
  if (idx >= tot) return;
  int e = idx & 15, lane = (idx >> 4) & 31;
  size_t tile = idx >> 9;
  int nt = (int)(tile % Ntiles), kt = (int)(tile / Ntiles);
  int n = nt * 16 + (lane & 15);
  int k = kt * 32 + e + 16 * (lane >> 4);
  float v = (k < Kih) ? wih[(size_t)n * Kih + k] : whh[(size_t)n * Khh + (k - Kih)];
  out[idx] = f2bf(v);
}
__global__ void wswz_ms(const float* __restrict__ mw, const float* __restrict__ sw,
                        __bf16* __restrict__ out) {  // K=1536, Np=96 (80 mel + stop)
  size_t idx = (size_t)blockIdx.x * blockDim.x + threadIdx.x;
  size_t tot = (size_t)48 * 6 * 512;
  if (idx >= tot) return;
  int e = idx & 15, lane = (idx >> 4) & 31;
  size_t tile = idx >> 9;
  int nt = (int)(tile % 6), kt = (int)(tile / 6);
  int n = nt * 16 + (lane & 15);
  int k = kt * 32 + e + 16 * (lane >> 4);
  float v = 0.f;
  if (n < 80) v = mw[(size_t)n * 1536 + k];
  else if (n == 80) v = sw[k];
  out[idx] = f2bf(v);
}

// ---- small utility kernels -------------------------------------------------
__global__ void zero_u32(uint32_t* p, int n) {
  int i = blockIdx.x * blockDim.x + threadIdx.x;
  if (i < n) p[i] = 0u;
}
__global__ void f32_to_bf16_k(const float* a, __bf16* o, size_t n) {
  size_t i = (size_t)blockIdx.x * blockDim.x + threadIdx.x;
  if (i < n) o[i] = f2bf(a[i]);
}
__global__ void vadd_k(const float* a, const float* b, float* o, int n) {
  int i = blockIdx.x * blockDim.x + threadIdx.x;
  if (i < n) o[i] = a[i] + b[i];
}
__global__ void make_msb_k(const float* mb, const float* sb, float* o) {
  int i = threadIdx.x;  // 96
  o[i] = (i < 80) ? mb[i] : (i == 80 ? sb[0] : 0.f);
}
__global__ void prevmel_k(const float* __restrict__ mel_tgt, __bf16* pmb, int t) {
  int i = blockIdx.x * blockDim.x + threadIdx.x;
  if (i >= B_ * 96) return;
  int b = i / 96, c = i % 96;
  float v = (t > 0 && c < MEL_) ? mel_tgt[((size_t)b * TD_ + (t - 1)) * MEL_ + c] : 0.f;
  pmb[i] = f2bf(v);
}
__global__ void locconv_k(const float* __restrict__ attprev, const float* __restrict__ alcw,
                          const float* __restrict__ alcb, __bf16* __restrict__ loc32) {
  int i = blockIdx.x * blockDim.x + threadIdx.x;
  if (i >= B_ * TE_ * 32) return;
  int c = i & 31, t = (i >> 5) & (TE_ - 1), b = i >> 14;
  float s = alcb[c];
  const float* w = alcw + c * 31;
  const float* ap = attprev + b * TE_;
  for (int j = 0; j < 31; ++j) {
    int tt = t + j - 15;
    if (tt >= 0 && tt < TE_) s += ap[tt] * w[j];
  }
  loc32[i] = f2bf(s);
}
__global__ void energies_k(const float* __restrict__ q, const float* __restrict__ memf,
                           const float* __restrict__ locp, const float* __restrict__ ab,
                           const float* __restrict__ ae, float* __restrict__ e) {
  int i = blockIdx.x * blockDim.x + threadIdx.x;
  if (i >= B_ * TE_) return;
  int b = i >> 9;
  const float* mp = memf + (size_t)i * ATT_;
  const float* lp = locp + (size_t)i * ATT_;
  const float* qp = q + b * ATT_;
  float s = 0.f;
  for (int c = 0; c < ATT_; ++c) s += ae[c] * tanhf(qp[c] + mp[c] + lp[c] + ab[c]);
  e[i] = s;
}
__global__ void softmax_k(const float* __restrict__ e, float* __restrict__ att,
                          float* __restrict__ attw, int tstep) {
  __shared__ float red[TE_];
  int b = blockIdx.x, tid = threadIdx.x;
  float v = e[b * TE_ + tid];
  red[tid] = v; __syncthreads();
  for (int s = TE_ / 2; s > 0; s >>= 1) {
    if (tid < s) red[tid] = fmaxf(red[tid], red[tid + s]);
    __syncthreads();
  }
  float mx = red[0]; __syncthreads();
  float ex = __expf(v - mx);
  red[tid] = ex; __syncthreads();
  for (int s = TE_ / 2; s > 0; s >>= 1) {
    if (tid < s) red[tid] += red[tid + s];
    __syncthreads();
  }
  float a = ex / red[0];
  att[b * TE_ + tid] = a;
  attw[((size_t)b * TD_ + tstep) * TE_ + tid] = a;
}
__global__ void ctx_k(const float* __restrict__ att, const float* __restrict__ enc,
                      __bf16* __restrict__ x0b, __bf16* __restrict__ pib) {
  int b = blockIdx.x, d = threadIdx.x;  // 512 threads
  const float* ep = enc + ((size_t)b * TE_) * ED_ + d;
  const float* ap = att + b * TE_;
  float s = 0.f;
  for (int t = 0; t < TE_; ++t) s += ap[t] * ep[(size_t)t * ED_];
  __bf16 hv = f2bf(s);
  x0b[(size_t)b * 1792 + 256 + d] = hv;
  pib[(size_t)b * 1536 + 1024 + d] = hv;
}
__global__ void lstm_pw_k(const float* __restrict__ gates, float* __restrict__ c,
                          __bf16* __restrict__ d1, int ld1,
                          __bf16* __restrict__ d2, int ld2) {
  int i = blockIdx.x * blockDim.x + threadIdx.x;
  if (i >= B_ * H_) return;
  int b = i >> 10, j = i & (H_ - 1);
  const float* g = gates + (size_t)b * 4 * H_;
  float iv = sigm(g[j]), fv = sigm(g[H_ + j]);
  float gv = tanhf(g[2 * H_ + j]), ov = sigm(g[3 * H_ + j]);
  float cn = fv * c[i] + iv * gv;
  c[i] = cn;
  __bf16 hb = f2bf(ov * tanhf(cn));
  d1[(size_t)b * ld1 + j] = hb;
  d2[(size_t)b * ld2 + j] = hb;
}
__global__ void scatter_ms_k(const float* __restrict__ ms, float* __restrict__ mel_out,
                             float* __restrict__ stop_out, int t) {
  int i = blockIdx.x * blockDim.x + threadIdx.x;
  if (i >= B_ * 81) return;
  int b = i / 81, c = i % 81;
  if (c < 80) mel_out[((size_t)b * TD_ + t) * MEL_ + c] = ms[(size_t)b * 96 + c];
  else        stop_out[(size_t)b * TD_ + t] = ms[(size_t)b * 96 + 80];
}
// postnet helpers
__global__ void post_fill0_k(const float* __restrict__ mel, __bf16* __restrict__ X) {
  size_t i = (size_t)blockIdx.x * blockDim.x + threadIdx.x;
  size_t tot = (size_t)B_ * 520 * 96;
  if (i >= tot) return;
  int c = (int)(i % 96);
  size_t rr = i / 96;
  int r = (int)(rr % 520), b = (int)(rr / 520), t = r - 2;
  float v = (t >= 0 && t < TD_ && c < MEL_) ? mel[((size_t)b * TD_ + t) * MEL_ + c] : 0.f;
  X[i] = f2bf(v);
}
__global__ void post_pw_k(const float* __restrict__ Y, int C, int ldy,
                          const float* __restrict__ g, const float* __restrict__ bt,
                          int dotanh, __bf16* __restrict__ Xn, int ldx) {
  size_t i = (size_t)blockIdx.x * blockDim.x + threadIdx.x;
  size_t tot = (size_t)B_ * 520 * ldx;
  if (i >= tot) return;
  int c = (int)(i % ldx);
  size_t rr = i / ldx;
  int r = (int)(rr % 520), b = (int)(rr / 520), t = r - 2;
  float v = 0.f;
  if (t >= 0 && t < TD_ && c < C) {
    float sc = g[c] / sqrtf(1.f + 1e-5f);
    v = Y[((size_t)b * 512 + t) * ldy + c] * sc + bt[c];
    if (dotanh) v = tanhf(v);
  }
  Xn[i] = f2bf(v);
}
__global__ void post_final_k(const float* __restrict__ Y, const float* __restrict__ g,
                             const float* __restrict__ bt, const float* __restrict__ mel,
                             float* __restrict__ out0) {
  size_t i = (size_t)blockIdx.x * blockDim.x + threadIdx.x;
  size_t tot = (size_t)B_ * TD_ * MEL_;
  if (i >= tot) return;
  int m = (int)(i % MEL_);
  size_t bt_ = i / MEL_;
  int t = (int)(bt_ % TD_), b = (int)(bt_ / TD_);
  float sc = g[m] / sqrtf(1.f + 1e-5f);
  float v = Y[((size_t)b * 512 + t) * 96 + m] * sc + bt[m];
  out0[i] = mel[i] + v;
}

// ---------------------------------------------------------------------------
extern "C" void kernel_launch(void* const* d_in, const int* in_sizes, int n_in,
                              void* d_out, int out_size, void* d_ws, size_t ws_size,
                              hipStream_t stream) {
  (void)in_sizes; (void)n_in; (void)out_size; (void)ws_size;
  const float* enc   = (const float*)d_in[0];
  const float* melt  = (const float*)d_in[1];
  const float* p1w   = (const float*)d_in[2];
  const float* p1b   = (const float*)d_in[3];
  const float* p2w   = (const float*)d_in[4];
  const float* p2b   = (const float*)d_in[5];
  const float* wih0  = (const float*)d_in[6];
  const float* whh0  = (const float*)d_in[7];
  const float* bih0  = (const float*)d_in[8];
  const float* bhh0  = (const float*)d_in[9];
  const float* wih1  = (const float*)d_in[10];
  const float* whh1  = (const float*)d_in[11];
  const float* bih1  = (const float*)d_in[12];
  const float* bhh1  = (const float*)d_in[13];
  const float* aq    = (const float*)d_in[14];
  const float* am    = (const float*)d_in[15];
  const float* al    = (const float*)d_in[16];
  const float* alcw  = (const float*)d_in[17];
  const float* alcb  = (const float*)d_in[18];
  const float* ae    = (const float*)d_in[19];
  const float* ab    = (const float*)d_in[20];
  const float* mw    = (const float*)d_in[21];
  const float* mb    = (const float*)d_in[22];
  const float* sw    = (const float*)d_in[23];
  const float* sb    = (const float*)d_in[24];
  const float* pw[5] = {(const float*)d_in[25], (const float*)d_in[29],
                        (const float*)d_in[33], (const float*)d_in[37],
                        (const float*)d_in[41]};
  const float* pb[5] = {(const float*)d_in[26], (const float*)d_in[30],
                        (const float*)d_in[34], (const float*)d_in[38],
                        (const float*)d_in[42]};
  const float* bg[5] = {(const float*)d_in[27], (const float*)d_in[31],
                        (const float*)d_in[35], (const float*)d_in[39],
                        (const float*)d_in[43]};
  const float* bb[5] = {(const float*)d_in[28], (const float*)d_in[32],
                        (const float*)d_in[36], (const float*)d_in[40],
                        (const float*)d_in[44]};

  float* out0     = (float*)d_out;                      // mel+post [B,TD,80]
  float* out_mel  = out0 + (size_t)B_ * TD_ * MEL_;     // mel
  float* out_stop = out_mel + (size_t)B_ * TD_ * MEL_;  // stop [B,TD,1]
  float* out_attw = out_stop + (size_t)B_ * TD_;        // attw [B,TD,TE]

  // ---- workspace carve-up ----
  char* wsb = (char*)d_ws;
  size_t cur = 0;
  auto alloc = [&](size_t bytes) -> void* {
    cur = (cur + 255) & ~(size_t)255;
    void* p = wsb + cur;
    cur += bytes;
    return p;
  };
  // bf16
  __bf16* encb  = (__bf16*)alloc((size_t)B_ * TE_ * ED_ * 2);
  __bf16* w0sw  = (__bf16*)alloc((size_t)56 * 256 * 512 * 2);
  __bf16* w1sw  = (__bf16*)alloc((size_t)64 * 256 * 512 * 2);
  __bf16* p1sw  = (__bf16*)alloc((size_t)3 * 16 * 512 * 2);
  __bf16* p2sw  = (__bf16*)alloc((size_t)8 * 16 * 512 * 2);
  __bf16* aqsw  = (__bf16*)alloc((size_t)32 * 8 * 512 * 2);
  __bf16* amsw  = (__bf16*)alloc((size_t)16 * 8 * 512 * 2);
  __bf16* alsw  = (__bf16*)alloc((size_t)1 * 8 * 512 * 2);
  __bf16* mssw  = (__bf16*)alloc((size_t)48 * 6 * 512 * 2);
  __bf16* pwsw[5];
  int pKt[5] = {3, 16, 16, 16, 16};
  int pNt[5] = {32, 32, 32, 32, 6};
  for (int l = 0; l < 5; ++l)
    pwsw[l] = (__bf16*)alloc((size_t)5 * pKt[l] * pNt[l] * 512 * 2);
  __bf16* pmb   = (__bf16*)alloc((size_t)B_ * 96 * 2);
  __bf16* pre1b = (__bf16*)alloc((size_t)B_ * 256 * 2);
  __bf16* x0b   = (__bf16*)alloc((size_t)B_ * 1792 * 2);
  __bf16* x1b   = (__bf16*)alloc((size_t)B_ * 2048 * 2);
  __bf16* pib   = (__bf16*)alloc((size_t)B_ * 1536 * 2);
  __bf16* loc32 = (__bf16*)alloc((size_t)B_ * TE_ * 32 * 2);
  __bf16* xA    = (__bf16*)alloc((size_t)B_ * 520 * 512 * 2);
  __bf16* xB    = (__bf16*)alloc((size_t)B_ * 520 * 512 * 2);
  // f32
  float* memf   = (float*)alloc((size_t)B_ * TE_ * ATT_ * 4);
  float* locp   = (float*)alloc((size_t)B_ * TE_ * ATT_ * 4);
  float* qb     = (float*)alloc((size_t)B_ * ATT_ * 4);
  float* eb     = (float*)alloc((size_t)B_ * TE_ * 4);
  float* attprev= (float*)alloc((size_t)B_ * TE_ * 4);
  float* gates  = (float*)alloc((size_t)B_ * 4 * H_ * 4);
  float* c0     = (float*)alloc((size_t)B_ * H_ * 4);
  float* c1     = (float*)alloc((size_t)B_ * H_ * 4);
  float* msf    = (float*)alloc((size_t)B_ * 96 * 4);
  float* b0     = (float*)alloc((size_t)4 * H_ * 4);
  float* b1     = (float*)alloc((size_t)4 * H_ * 4);
  float* msb    = (float*)alloc((size_t)96 * 4);
  float* yb     = (float*)alloc((size_t)B_ * 512 * 512 * 4);

  const int TB = 256;
  auto blocks = [](size_t n, int tb) { return (unsigned)((n + tb - 1) / tb); };

  // ---- init: zero recurrent state ----
  zero_u32<<<blocks(B_ * TE_, TB), TB, 0, stream>>>((uint32_t*)attprev, B_ * TE_);
  zero_u32<<<blocks(B_ * H_, TB), TB, 0, stream>>>((uint32_t*)c0, B_ * H_);
  zero_u32<<<blocks(B_ * H_, TB), TB, 0, stream>>>((uint32_t*)c1, B_ * H_);
  zero_u32<<<blocks(B_ * 1792 / 2, TB), TB, 0, stream>>>((uint32_t*)x0b, B_ * 1792 / 2);
  zero_u32<<<blocks(B_ * 2048 / 2, TB), TB, 0, stream>>>((uint32_t*)x1b, B_ * 2048 / 2);
  zero_u32<<<blocks(B_ * 1536 / 2, TB), TB, 0, stream>>>((uint32_t*)pib, B_ * 1536 / 2);

  // ---- weight conversion ----
  {
    size_t n = (size_t)B_ * TE_ * ED_;
    f32_to_bf16_k<<<blocks(n, TB), TB, 0, stream>>>(enc, encb, n);
  }
  auto sw_launch = [&](const float* W, int N, int K, int sN, int sK, __bf16* o,
                       int Kt, int Nt) {
    size_t n = (size_t)Kt * Nt * 512;
    wswz<<<blocks(n, TB), TB, 0, stream>>>(W, N, K, sN, sK, o, Kt, Nt);
  };
  sw_launch(p1w, 256, 80, 80, 1, p1sw, 3, 16);
  sw_launch(p2w, 256, 256, 256, 1, p2sw, 8, 16);
  sw_launch(aq, 128, 1024, 1024, 1, aqsw, 32, 8);
  sw_launch(am, 128, 512, 512, 1, amsw, 16, 8);
  sw_launch(al, 128, 32, 32, 1, alsw, 1, 8);
  {
    size_t n = (size_t)56 * 256 * 512;
    wswz_cat<<<blocks(n, TB), TB, 0, stream>>>(wih0, whh0, 768, 1024, w0sw, 56, 256);
    n = (size_t)64 * 256 * 512;
    wswz_cat<<<blocks(n, TB), TB, 0, stream>>>(wih1, whh1, 1024, 1024, w1sw, 64, 256);
    n = (size_t)48 * 6 * 512;
    wswz_ms<<<blocks(n, TB), TB, 0, stream>>>(mw, sw, mssw);
  }
  int pCin[5] = {80, 512, 512, 512, 512};
  int pCout[5] = {512, 512, 512, 512, 80};
  for (int l = 0; l < 5; ++l)
    for (int k = 0; k < 5; ++k)
      sw_launch(pw[l] + k, pCout[l], pCin[l], pCin[l] * 5, 5,
                pwsw[l] + (size_t)k * pKt[l] * pNt[l] * 512, pKt[l], pNt[l]);
  vadd_k<<<blocks(4 * H_, TB), TB, 0, stream>>>(bih0, bhh0, b0, 4 * H_);
  vadd_k<<<blocks(4 * H_, TB), TB, 0, stream>>>(bih1, bhh1, b1, 4 * H_);
  make_msb_k<<<1, 96, 0, stream>>>(mb, sb, msb);

  // mem = enc @ am.T   [32768,512]x[512,128]
  gemm_ws<4><<<dim3(2, 2048), 32, 0, stream>>>(
      encb, ED_, amsw, (const float*)nullptr, memf, ATT_,
      (__bf16*)nullptr, 0, 16, 8, 0);

  // ---- sequential decoder ----
  for (int t = 0; t < TD_; ++t) {
    prevmel_k<<<blocks(B_ * 96, TB), TB, 0, stream>>>(melt, pmb, t);
    // prenet
    gemm_ws<4><<<dim3(4, 4), 32, 0, stream>>>(
        pmb, 96, p1sw, p1b, (float*)nullptr, 0, pre1b, 256, 3, 16, 1);
    gemm_ws<4><<<dim3(4, 4), 32, 0, stream>>>(
        pre1b, 256, p2sw, p2b, (float*)nullptr, 0, x0b, 1792, 8, 16, 1);
    // location attention
    locconv_k<<<blocks((size_t)B_ * TE_ * 32, TB), TB, 0, stream>>>(attprev, alcw, alcb, loc32);
    gemm_ws<4><<<dim3(2, 2048), 32, 0, stream>>>(
        loc32, 32, alsw, (const float*)nullptr, locp, ATT_,
        (__bf16*)nullptr, 0, 1, 8, 0);
    gemm_ws<4><<<dim3(2, 4), 32, 0, stream>>>(
        pib, 1536, aqsw, (const float*)nullptr, qb, ATT_,
        (__bf16*)nullptr, 0, 32, 8, 0);
    energies_k<<<blocks(B_ * TE_, TB), TB, 0, stream>>>(qb, memf, locp, ab, ae, eb);
    softmax_k<<<B_, TE_, 0, stream>>>(eb, attprev, out_attw, t);
    ctx_k<<<B_, ED_, 0, stream>>>(attprev, enc, x0b, pib);
    // LSTM layer 0
    gemm_ws<4><<<dim3(64, 4), 32, 0, stream>>>(
        x0b, 1792, w0sw, b0, gates, 4 * H_, (__bf16*)nullptr, 0, 56, 256, 0);
    lstm_pw_k<<<blocks(B_ * H_, TB), TB, 0, stream>>>(gates, c0, x1b, 2048, x0b + 768, 1792);
    // LSTM layer 1
    gemm_ws<4><<<dim3(64, 4), 32, 0, stream>>>(
        x1b, 2048, w1sw, b1, gates, 4 * H_, (__bf16*)nullptr, 0, 64, 256, 0);
    lstm_pw_k<<<blocks(B_ * H_, TB), TB, 0, stream>>>(gates, c1, pib, 1536, x1b + 1024, 2048);
    // mel + stop projection (combined N=96)
    gemm_ws<2><<<dim3(3, 4), 32, 0, stream>>>(
        pib, 1536, mssw, msb, msf, 96, (__bf16*)nullptr, 0, 48, 6, 0);
    scatter_ms_k<<<blocks(B_ * 81, TB), TB, 0, stream>>>(msf, out_mel, out_stop, t);
  }

  // ---- postnet ----
  {
    size_t n0 = (size_t)B_ * 520 * 96;
    post_fill0_k<<<blocks(n0, TB), TB, 0, stream>>>(out_mel, xA);
  }
  size_t npw = (size_t)B_ * 520 * 512;
  // L0: xA(96) -> y -> xB
  gemm_conv<4><<<dim3(8, B_ * 16), 32, 0, stream>>>(xA, 96, pwsw[0], pb[0], yb, 512, 3, 32);
  post_pw_k<<<blocks(npw, TB), TB, 0, stream>>>(yb, 512, 512, bg[0], bb[0], 1, xB, 512);
  // L1: xB -> xA
  gemm_conv<4><<<dim3(8, B_ * 16), 32, 0, stream>>>(xB, 512, pwsw[1], pb[1], yb, 512, 16, 32);
  post_pw_k<<<blocks(npw, TB), TB, 0, stream>>>(yb, 512, 512, bg[1], bb[1], 1, xA, 512);
  // L2: xA -> xB
  gemm_conv<4><<<dim3(8, B_ * 16), 32, 0, stream>>>(xA, 512, pwsw[2], pb[2], yb, 512, 16, 32);
  post_pw_k<<<blocks(npw, TB), TB, 0, stream>>>(yb, 512, 512, bg[2], bb[2], 1, xB, 512);
  // L3: xB -> xA
  gemm_conv<4><<<dim3(8, B_ * 16), 32, 0, stream>>>(xB, 512, pwsw[3], pb[3], yb, 512, 16, 32);
  post_pw_k<<<blocks(npw, TB), TB, 0, stream>>>(yb, 512, 512, bg[3], bb[3], 1, xA, 512);
  // L4: xA -> y(96) -> out0 = mel + bn(y)
  gemm_conv<2><<<dim3(3, B_ * 16), 32, 0, stream>>>(xA, 512, pwsw[4], pb[4], yb, 96, 16, 6);
  {
    size_t n = (size_t)B_ * TD_ * MEL_;
    post_final_k<<<blocks(n, TB), TB, 0, stream>>>(yb, bg[4], bb[4], out_mel, out0);
  }
}
